// PI_v5_45664092291700
// MI455X (gfx1250) — compile-verified, matrix operand order checked by reading, not run on hardware
//
#include <hip/hip_runtime.h>

// ---------------------------------------------------------------------------
// CDNA5 / gfx1250 implementation.
// Pipeline:
//   prep_bfrag     : pack W_sub/W_obj/W_reg/W_fc(permuted) into per-lane
//                    v_wmma_f32_16x16x32_f16 B-fragment layout, split hi/lo f16
//   gemm_wmma<0>   : sub_all / obj_all = relu(feature_obj) @ W + b
//   gemm_wmma<1>   : feat_reg[n][hw*512+m] = relu(region) @ W_reg + b_reg
//   gemm_wmma<2>   : out = relu(feat_reg[idx_r] + sub[idx_s] + obj[idx_o]) @ Wfc' + b_fc
//                    (gather + fuse + relu done in-register while building A frags)
// Precision: split-f16 (hi + lo) emulation, 3 WMMAs per 16x16x32 step,
// fp32 accumulate -> ~fp32 accuracy at 16-bit matrix-core rates.
// Each wave register-blocks 8 N-tiles (16x128 strip) so the split-f16 A
// construction (the VALU-heavy part) is amortized over 24 WMMAs per K-chunk.
// ---------------------------------------------------------------------------

typedef __attribute__((ext_vector_type(16))) _Float16 v16h;
typedef __attribute__((ext_vector_type(8)))  float    v8f;

constexpr int DIM    = 512;
constexpr int POOL2  = 49;             // 7*7
constexpr int KFC    = DIM * POOL2;    // 25088
constexpr int NTN    = DIM / 16;       // 32 N-tiles (N == 512 for every GEMM here)
constexpr int NTW    = 8;              // N-tiles per wave (16x128 strip)
constexpr int NGRP   = NTN / NTW;      // 4 wave-column groups

// ---------------------------------------------------------------------------
// Pack a (K x 512) fp32 weight matrix into WMMA B-fragment layout.
// Per (n-tile jt, k-chunk kc): block of 1024 halves:
//   [0..511]    hi frags, lane-major: lane*16 + e
//   [512..1023] lo frags
// Per-lane element e maps to K = kc*32 + klocal(g,e), per ISA 16-bit B layout:
//   lanes 0-15 hold K 0-7 / 16-23, lanes 16-31 hold K 8-15 / 24-31.
// permute==1: K is reindexed k' = hw*512 + m  ->  original row m*49 + hw (W_fc).
// ---------------------------------------------------------------------------
__global__ __launch_bounds__(256) void prep_bfrag(const float* __restrict__ W,
                                                  _Float16* __restrict__ frag,
                                                  int nkc, int permute) {
  int tid  = blockIdx.x * blockDim.x + threadIdx.x;
  int lane = tid & 31;
  int rest = tid >> 5;
  if (rest >= nkc * NTN) return;
  int kc = rest % nkc;
  int jt = rest / nkc;
  int n16 = lane & 15;
  int g   = lane >> 4;
  int j   = jt * 16 + n16;
  _Float16* blk = frag + (size_t)(jt * nkc + kc) * 1024;
#pragma unroll
  for (int e = 0; e < 16; ++e) {
    int klocal = (e < 8) ? (8 * g + e) : (16 + 8 * g + (e - 8));
    int k = kc * 32 + klocal;
    int korig = permute ? ((k & (DIM - 1)) * POOL2 + (k >> 9)) : k;
    float w = W[(size_t)korig * DIM + j];
    _Float16 h = (_Float16)w;
    blk[lane * 16 + e]       = h;
    blk[512 + lane * 16 + e] = (_Float16)(w - (float)h);
  }
}

// ---------------------------------------------------------------------------
// Generic WMMA GEMM. One wave owns a 16(M) x 128(N) strip = 8 accumulators.
// K-loop in chunks of 32 with split-f16 A built once per chunk.
// MODE 0: A = relu(feature_obj[row])                (K = 512)
// MODE 1: A = relu(region[n][c][hw]), row = n*49+hw (K = 512, stride-49 reads)
// MODE 2: A = relu(reg_row[k'] + sub[m] + obj[m])   (K = 25088, fused gather)
// ---------------------------------------------------------------------------
template <int MODE>
__global__ __launch_bounds__(256) void gemm_wmma(
    const float* __restrict__ A0,   // m0: feature_obj  m1: feature_region  m2: feat_reg
    const float* __restrict__ A1,   // m2: sub_all
    const float* __restrict__ A2,   // m2: obj_all
    const int*   __restrict__ pred, // m2: mat_predicate (Nx3)
    const _Float16* __restrict__ bfrag,
    const float* __restrict__ bias,
    float* __restrict__ out,
    int mtiles, int nkc) {
  int wid = blockIdx.x * (blockDim.x >> 5) + (threadIdx.x >> 5);
  int mt  = wid / NGRP;
  int ng  = wid % NGRP;
  int nt0 = ng * NTW;
  if (mt >= mtiles) return;            // wave-uniform: EXEC stays all-1s for WMMA

  int lane = threadIdx.x & 31;
  int n16  = lane & 15;
  int g    = lane >> 4;
  int row  = mt * 16 + n16;            // A row owned by this lane (both g groups)

  const float* ra = nullptr;
  const float* rs = nullptr;
  const float* ro = nullptr;
  if constexpr (MODE == 0) {
    ra = A0 + (size_t)row * DIM;
  } else if constexpr (MODE == 1) {
    int n  = row / POOL2;
    int hw = row % POOL2;
    ra = A0 + (size_t)n * KFC + hw;    // element c at +c*49
  } else {
    int is = pred[row * 3 + 0];
    int io = pred[row * 3 + 1];
    int ir = pred[row * 3 + 2];
    ra = A0 + (size_t)ir * KFC;
    rs = A1 + (size_t)is * DIM;
    ro = A2 + (size_t)io * DIM;
  }

  v8f c[NTW];
#pragma unroll
  for (int j = 0; j < NTW; ++j) c[j] = (v8f){};

  for (int kc = 0; kc < nkc; ++kc) {
    int kb = kc * 32;
    float va[16];

    if constexpr (MODE == 1) {
#pragma unroll
      for (int e = 0; e < 16; ++e) {
        int klocal = (e < 8) ? (8 * g + e) : (16 + 8 * g + (e - 8));
        float x = ra[(size_t)(kb + klocal) * POOL2];
        va[e] = x > 0.f ? x : 0.f;
      }
      if (kc + 1 < nkc)                       // lookahead into L2 for the stream
        __builtin_prefetch(ra + (size_t)(kb + 32) * POOL2, 0, 1);
    } else {
#pragma unroll
      for (int r = 0; r < 2; ++r) {
        int ks = kb + 8 * g + r * 16;  // 8-aligned; runs never cross 512 boundary
        const float4* pa = reinterpret_cast<const float4*>(ra + ks);
        float4 xa = pa[0], xb = pa[1];
        float t[8] = {xa.x, xa.y, xa.z, xa.w, xb.x, xb.y, xb.z, xb.w};
        if constexpr (MODE == 2) {
          int m0 = ks & (DIM - 1);
          const float4* ps = reinterpret_cast<const float4*>(rs + m0);
          const float4* po = reinterpret_cast<const float4*>(ro + m0);
          float4 sa = ps[0], sb = ps[1], oa = po[0], ob = po[1];
          t[0] += sa.x + oa.x; t[1] += sa.y + oa.y;
          t[2] += sa.z + oa.z; t[3] += sa.w + oa.w;
          t[4] += sb.x + ob.x; t[5] += sb.y + ob.y;
          t[6] += sb.z + ob.z; t[7] += sb.w + ob.w;
        }
#pragma unroll
        for (int e = 0; e < 8; ++e) {
          float x = t[e];
          va[r * 8 + e] = x > 0.f ? x : 0.f;
        }
      }
      if (kc + 1 < nkc)
        __builtin_prefetch(ra + kb + 32, 0, 1);
    }

    // split-f16 A fragments (built ONCE per K-chunk, reused by 8 N-tiles)
    v16h ahi, alo;
#pragma unroll
    for (int e = 0; e < 16; ++e) {
      _Float16 h = (_Float16)va[e];
      ahi[e] = h;
      alo[e] = (_Float16)(va[e] - (float)h);
    }

    // stream 8 prepacked B fragment pairs; 3 WMMAs each (hi*hi, hi*lo, lo*hi)
#pragma unroll
    for (int j = 0; j < NTW; ++j) {
      const _Float16* blk = bfrag + (size_t)((nt0 + j) * nkc + kc) * 1024;
      v16h bhi = *reinterpret_cast<const v16h*>(blk + lane * 16);
      v16h blo = *reinterpret_cast<const v16h*>(blk + 512 + lane * 16);
      c[j] = __builtin_amdgcn_wmma_f32_16x16x32_f16(false, ahi, false, bhi, (short)0, c[j], false, false);
      c[j] = __builtin_amdgcn_wmma_f32_16x16x32_f16(false, ahi, false, blo, (short)0, c[j], false, false);
      c[j] = __builtin_amdgcn_wmma_f32_16x16x32_f16(false, alo, false, bhi, (short)0, c[j], false, false);
    }
  }

  // epilogue: D element i of tile j -> (M = mt*16 + 8*g + i, N = (nt0+j)*16 + n16)
#pragma unroll
  for (int j = 0; j < NTW; ++j) {
    int ncol = (nt0 + j) * 16 + n16;
    float bv = bias[ncol];
#pragma unroll
    for (int i = 0; i < 8; ++i) {
      int m = mt * 16 + g * 8 + i;
      out[(size_t)m * DIM + ncol] = c[j][i] + bv;
    }
  }
}

extern "C" void kernel_launch(void* const* d_in, const int* in_sizes, int n_in,
                              void* d_out, int out_size, void* d_ws, size_t ws_size,
                              hipStream_t stream) {
  const float* feature_obj    = (const float*)d_in[0];
  const float* feature_region = (const float*)d_in[1];
  const int*   mat_pred       = (const int*)  d_in[2];
  const float* W_sub = (const float*)d_in[3];
  const float* b_sub = (const float*)d_in[4];
  const float* W_obj = (const float*)d_in[5];
  const float* b_obj = (const float*)d_in[6];
  const float* W_reg = (const float*)d_in[7];
  const float* b_reg = (const float*)d_in[8];
  const float* W_fc  = (const float*)d_in[9];
  const float* b_fc  = (const float*)d_in[10];
  (void)in_sizes; (void)n_in; (void)out_size; (void)ws_size;

  char* ws = (char*)d_ws;
  size_t off = 0;
  auto alloc = [&](size_t bytes) -> void* {
    off = (off + 255) & ~(size_t)255;
    void* p = ws + off;
    off += bytes;
    return p;
  };

  constexpr int N_OBJ  = 2048;
  constexpr int N_REG  = 4096;
  constexpr int N_PRED = 8192;
  constexpr int NKC_S  = DIM / 32;   // 16
  constexpr int NKC_FC = KFC / 32;   // 784

  float*     sub_all  = (float*)    alloc((size_t)N_OBJ * DIM * 4);
  float*     obj_all  = (float*)    alloc((size_t)N_OBJ * DIM * 4);
  float*     feat_reg = (float*)    alloc((size_t)N_REG * KFC * 4);
  _Float16*  frag_sub = (_Float16*) alloc((size_t)NTN * NKC_S  * 1024 * 2);
  _Float16*  frag_obj = (_Float16*) alloc((size_t)NTN * NKC_S  * 1024 * 2);
  _Float16*  frag_reg = (_Float16*) alloc((size_t)NTN * NKC_S  * 1024 * 2);
  _Float16*  frag_fc  = (_Float16*) alloc((size_t)NTN * NKC_FC * 1024 * 2);

  // --- pack B fragments (hi/lo split f16, per-lane WMMA layout) ---
  {
    int t  = NTN * NKC_S * 32;
    int tb = (t + 255) / 256;
    prep_bfrag<<<tb, 256, 0, stream>>>(W_sub, frag_sub, NKC_S, 0);
    prep_bfrag<<<tb, 256, 0, stream>>>(W_obj, frag_obj, NKC_S, 0);
    prep_bfrag<<<tb, 256, 0, stream>>>(W_reg, frag_reg, NKC_S, 0);
    int tf  = NTN * NKC_FC * 32;
    int tfb = (tf + 255) / 256;
    prep_bfrag<<<tfb, 256, 0, stream>>>(W_fc, frag_fc, NKC_FC, 1);
  }

  // --- stage 1: object projections ---
  {
    int mtiles = N_OBJ / 16;                 // 128
    int waves  = mtiles * NGRP;              // 512
    int blocks = (waves + 7) / 8;
    gemm_wmma<0><<<blocks, 256, 0, stream>>>(feature_obj, nullptr, nullptr, nullptr,
                                             frag_sub, b_sub, sub_all, mtiles, NKC_S);
    gemm_wmma<0><<<blocks, 256, 0, stream>>>(feature_obj, nullptr, nullptr, nullptr,
                                             frag_obj, b_obj, obj_all, mtiles, NKC_S);
  }

  // --- stage 2: region projection (4096*49 rows) ---
  {
    int mtiles = (N_REG * POOL2) / 16;       // 12544
    int waves  = mtiles * NGRP;              // 50176
    int blocks = (waves + 7) / 8;
    gemm_wmma<1><<<blocks, 256, 0, stream>>>(feature_region, nullptr, nullptr, nullptr,
                                             frag_reg, b_reg, feat_reg, mtiles, NKC_S);
  }

  // --- stage 3: fused gather + relu + FC GEMM ---
  {
    int mtiles = N_PRED / 16;                // 512
    int waves  = mtiles * NGRP;              // 2048
    int blocks = (waves + 7) / 8;
    gemm_wmma<2><<<blocks, 256, 0, stream>>>(feat_reg, sub_all, obj_all, mat_pred,
                                             frag_fc, b_fc, (float*)d_out, mtiles, NKC_FC);
  }
}
